// GraphAttention_48387101557119
// MI455X (gfx1250) — matrix operand order repaired and checked by reading, not compiled, not run on hardware
//
#include <hip/hip_runtime.h>

typedef __attribute__((ext_vector_type(16))) __bf16 v16bf;
typedef __attribute__((ext_vector_type(8)))  __bf16 v8bf;
typedef __attribute__((ext_vector_type(8)))  float  v8f;

#define BM 128
#define BN 128
#define BK 32
#define THREADS 256
#define LDK (BK + 8)   // 80B row stride: 16B-aligned chunks, min-conflict b128 frags

enum { EPI_STORE = 0, EPI_BIAS = 1, EPI_MASK_TANH = 2, EPI_RELU = 3,
       EPI_F32_BF16 = 4, EPI_GATE = 5 };

struct GemmP {
  const __bf16* A; long long sAb, sAh; int lda;
  const __bf16* B; long long sBb, sBh; int ldb;
  __bf16* Cb; long long sCbB, sCbH; int ldcb;
  float* Cf; long long sCfB; int ldcf;
  const float* bias; long long sBiasH;
  const unsigned char* mask; long long sMaskB; int ldm;
  const float* hf; const float* ha; float* outp; const float* gbias;
  int M, Ncols, K, H, epi;
};

// ---------------------------------------------------------------------------
// Batched bf16 GEMM, C = epilogue(A @ B + ...)
//   A: row-major [M x K] bf16
//   B: n-major Bsrc[n][k] (each output column's K-vector contiguous; this is
//      the natural layout of weight[out,in] and of pre-transposed activations)
// Tile 128x128x32, 8 wave32s in 2x4; each wave 64x32 = 8 x v_wmma 16x16x32.
// M, Ncols multiples of 128; K multiple of 32 (true for all launches here).
// ---------------------------------------------------------------------------
__global__ __launch_bounds__(THREADS) void gat_gemm(GemmP p) {
  __shared__ __bf16 lA[2][BM][LDK];
  __shared__ __bf16 lB[2][BN][LDK];

  const int tid  = threadIdx.x;
  const int lane = tid & 31;
  const int wave = tid >> 5;
  const int waveM = (wave >> 2) * 64;   // 0 / 64
  const int waveN = (wave & 3) * 32;    // 0 / 32 / 64 / 96
  const int bMi = blockIdx.y, bNi = blockIdx.x, z = blockIdx.z;
  const int b  = z / p.H;
  const int hd = z - b * p.H;

  const __bf16* A  = p.A + b * p.sAb + hd * p.sAh + (size_t)bMi * BM * p.lda;
  const __bf16* Bp = p.B + b * p.sBb + hd * p.sBh + (size_t)bNi * BN * p.ldb;

  // global->LDS mapping: 256 threads move 128x32 bf16 (8KB) per operand
  const int fr = tid >> 2;          // 0..63 (two rows: fr, fr+64)
  const int fc = (tid & 3) * 8;     // 0,8,16,24

  const int S = p.K / BK;

  v8bf a0, a1, b0, b1;
  auto fetch = [&](int s) {
    const long long k0 = (long long)s * BK;
    a0 = *(const v8bf*)(A  + (size_t)fr        * p.lda + k0 + fc);
    a1 = *(const v8bf*)(A  + (size_t)(fr + 64) * p.lda + k0 + fc);
    b0 = *(const v8bf*)(Bp + (size_t)fr        * p.ldb + k0 + fc);
    b1 = *(const v8bf*)(Bp + (size_t)(fr + 64) * p.ldb + k0 + fc);
    // prefetch 2 stages ahead (one lane per 64B row) -> global_prefetch_b8
    if (s + 2 < S && (tid & 3) == 0) {
      const long long k2 = k0 + 2 * BK;
      __builtin_prefetch(A  + (size_t)fr        * p.lda + k2, 0, 1);
      __builtin_prefetch(A  + (size_t)(fr + 64) * p.lda + k2, 0, 1);
      __builtin_prefetch(Bp + (size_t)fr        * p.ldb + k2, 0, 1);
      __builtin_prefetch(Bp + (size_t)(fr + 64) * p.ldb + k2, 0, 1);
    }
  };
  auto stash = [&](int buf) {
    *(v8bf*)&lA[buf][fr][fc]      = a0;
    *(v8bf*)&lA[buf][fr + 64][fc] = a1;
    *(v8bf*)&lB[buf][fr][fc]      = b0;
    *(v8bf*)&lB[buf][fr + 64][fc] = b1;
  };

  v8f acc[4][2];
#pragma unroll
  for (int mt = 0; mt < 4; ++mt)
#pragma unroll
    for (int nt = 0; nt < 2; ++nt)
#pragma unroll
      for (int i = 0; i < 8; ++i) acc[mt][nt][i] = 0.f;

  const int lm = lane & 15;
  const int ka = (lane < 16) ? 0 : 8;    // A frag: K ka..ka+7 and ka+16..ka+23
  const int kb = (lane < 16) ? 0 : 8;    // B frag: K kb..kb+7 and kb+16..kb+23

  auto mma = [&](int buf) {
    v16bf af[4], bf[2];
#pragma unroll
    for (int nt = 0; nt < 2; ++nt) {
      const int n = waveN + nt * 16 + lm;
      v8bf lo = *(const v8bf*)&lB[buf][n][kb];
      v8bf hi = *(const v8bf*)&lB[buf][n][kb + 16];
#pragma unroll
      for (int i = 0; i < 8; ++i) { bf[nt][i] = lo[i]; bf[nt][i + 8] = hi[i]; }
    }
#pragma unroll
    for (int mt = 0; mt < 4; ++mt) {
      const int m = waveM + mt * 16 + lm;
      v8bf lo = *(const v8bf*)&lA[buf][m][ka];
      v8bf hi = *(const v8bf*)&lA[buf][m][ka + 16];
#pragma unroll
      for (int i = 0; i < 8; ++i) { af[mt][i] = lo[i]; af[mt][i + 8] = hi[i]; }
    }
#pragma unroll
    for (int mt = 0; mt < 4; ++mt)
#pragma unroll
      for (int nt = 0; nt < 2; ++nt)
        acc[mt][nt] = __builtin_amdgcn_wmma_f32_16x16x32_bf16(
            false, af[mt], false, bf[nt], (short)0, acc[mt][nt], false, false);
  };

  fetch(0); stash(0);
  __syncthreads();
  for (int s = 0; s < S; ++s) {
    if (s + 1 < S) fetch(s + 1);   // global loads overlap compute
    mma(s & 1);
    if (s + 1 < S) stash((s + 1) & 1);
    __syncthreads();
  }

  // ---- fused epilogue ----
  __bf16* Cb = p.Cb ? p.Cb + b * p.sCbB + hd * p.sCbH : nullptr;
  float*  Cf = p.Cf ? p.Cf + b * p.sCfB : nullptr;
  const float* bias = p.bias ? p.bias + hd * p.sBiasH : nullptr;
  const unsigned char* mask = p.mask ? p.mask + b * p.sMaskB : nullptr;
  const size_t zo = (size_t)b * p.M * p.Ncols;

  const int mbase = bMi * BM + waveM + ((lane >> 4) << 3);
  const int nbase = bNi * BN + waveN + lm;

#pragma unroll
  for (int mt = 0; mt < 4; ++mt) {
#pragma unroll
    for (int nt = 0; nt < 2; ++nt) {
      const int gn = nbase + nt * 16;
#pragma unroll
      for (int r = 0; r < 8; ++r) {
        const int gm = mbase + mt * 16 + r;
        float v = acc[mt][nt][r];
        switch (p.epi) {
          case EPI_BIAS:
            v += bias[gn];
            Cb[(size_t)gm * p.ldcb + gn] = (__bf16)v;
            break;
          case EPI_STORE:
            Cb[(size_t)gm * p.ldcb + gn] = (__bf16)v;
            break;
          case EPI_RELU:
            v = v > 0.f ? v : 0.f;
            Cb[(size_t)gm * p.ldcb + gn] = (__bf16)v;
            break;
          case EPI_MASK_TANH: {
            const unsigned char mk = mask[(size_t)gm * p.ldm + gn];
            float t = 0.f;
            if (mk) {
              const float ax = fabsf(v);
              const float e  = exp2f(-2.885390082f * ax);  // exp(-2|v|)
              t = (1.f - e) / (1.f + e);
              t = (v < 0.f) ? -t : t;
            }
            Cb[(size_t)gm * p.ldcb + gn] = (__bf16)t;
          } break;
          case EPI_F32_BF16:
            Cf[(size_t)gm * p.ldcf + gn] = v;
            Cb[(size_t)gm * p.ldcb + gn] = (__bf16)v;
            break;
          case EPI_GATE: {
            const float g  = v + p.gbias[gn];
            const float sg = 1.f / (1.f + exp2f(-1.442695041f * g));
            const size_t idx = zo + (size_t)gm * p.Ncols + gn;
            const float hfv = p.hf[idx], hav = p.ha[idx];
            p.outp[idx] = hfv * sg + hav * (1.f - sg);
          } break;
        }
      }
    }
  }
}

// ---- prep / transpose kernels ----------------------------------------------
__global__ void k_cvt(const float* __restrict__ s, __bf16* __restrict__ d,
                      long long n) {
  long long i  = (long long)blockIdx.x * blockDim.x + threadIdx.x;
  long long st = (long long)gridDim.x * blockDim.x;
  for (; i < n; i += st) d[i] = (__bf16)s[i];
}

__global__ void k_attnT(const float* __restrict__ s, __bf16* __restrict__ d) {
  // attn [4][512][512] -> attnT[hd][p][o] bf16 (n-major B operand)
  const long long i = (long long)blockIdx.x * blockDim.x + threadIdx.x;
  const int hd = (int)(i >> 18);
  const int rem = (int)(i & 262143);
  const int o = rem >> 9, pc = rem & 511;
  d[((long long)hd << 18) + ((long long)pc << 9) + o] = (__bf16)s[i];
}

__global__ void k_gcat(const float* __restrict__ g1, const float* __restrict__ g2,
                       __bf16* __restrict__ d) {
  // gcat[o][k], k<512 from g1[o][k], else g2[o][k-512]  (K=1024 gate weight)
  const long long i = (long long)blockIdx.x * blockDim.x + threadIdx.x;
  const int o = (int)(i >> 10), k = (int)(i & 1023);
  d[i] = (__bf16)((k < 512) ? g1[o * 512 + k] : g2[o * 512 + (k - 512)]);
}

// Wh [z][1024 m][512 o] -> WhT [z][512 o][1024 m], 64x64 LDS tiles, b128 both sides
__global__ __launch_bounds__(256) void k_transpose(const __bf16* __restrict__ src,
                                                   __bf16* __restrict__ dst) {
  __shared__ __bf16 lT[64][72];   // 144B row stride: 16B-aligned, bank-spread
  const long long zo = (long long)blockIdx.z * 1024 * 512;
  const int m0 = blockIdx.y * 64, o0 = blockIdx.x * 64;
  const int t = threadIdx.x;
  const int r = t >> 3, c = (t & 7) * 8;   // r: 0..31, c: 0..56
  const __bf16* s0 = src + zo + (size_t)(m0 + r) * 512 + o0 + c;
  v8bf v0 = *(const v8bf*)s0;
  v8bf v1 = *(const v8bf*)(s0 + (size_t)32 * 512);
#pragma unroll
  for (int j = 0; j < 8; ++j) { lT[c + j][r] = v0[j]; lT[c + j][r + 32] = v1[j]; }
  __syncthreads();
  __bf16* d0 = dst + zo + (size_t)(o0 + r) * 1024 + m0 + c;
  *(v8bf*)d0                       = *(const v8bf*)&lT[r][c];
  *(v8bf*)(d0 + (size_t)32 * 1024) = *(const v8bf*)&lT[r + 32][c];
}

// ---- host -------------------------------------------------------------------
extern "C" void kernel_launch(void* const* d_in, const int* in_sizes, int n_in,
                              void* d_out, int out_size, void* d_ws, size_t ws_size,
                              hipStream_t stream) {
  (void)in_sizes; (void)n_in; (void)out_size; (void)ws_size;
  const float*         h    = (const float*)d_in[0];
  const unsigned char* adj  = (const unsigned char*)d_in[1];  // bool, 1B/elem
  const float*         W    = (const float*)d_in[2];
  const float*         bW   = (const float*)d_in[3];
  const float*         attn = (const float*)d_in[4];
  const float*         A_w  = (const float*)d_in[5];
  const float*         fc_w = (const float*)d_in[6];
  const float*         g1   = (const float*)d_in[7];
  const float*         g2   = (const float*)d_in[8];
  const float*         gb   = (const float*)d_in[9];
  float* out = (float*)d_out;

  constexpr int Bn = 16, Nn = 1024, IN = 512, OUT = 512, Hn = 4;

  char* ws = (char*)d_ws; size_t off = 0;
  auto alloc = [&](size_t bytes) -> char* {
    char* r = ws + off; off += (bytes + 255) & ~(size_t)255; return r;
  };
  __bf16* h_bf   = (__bf16*)alloc((size_t)Bn * Nn * IN * 2);        // 16.8 MB
  __bf16* W_bf   = (__bf16*)alloc((size_t)Hn * OUT * IN * 2);
  __bf16* at_bf  = (__bf16*)alloc((size_t)Hn * OUT * OUT * 2);
  __bf16* Aw_bf  = (__bf16*)alloc((size_t)OUT * Hn * OUT * 2);
  __bf16* fc_bf  = (__bf16*)alloc((size_t)OUT * IN * 2);
  __bf16* gc_bf  = (__bf16*)alloc((size_t)OUT * 2 * OUT * 2);
  __bf16* Wh_bf  = (__bf16*)alloc((size_t)Bn * Hn * Nn * OUT * 2);  // 67 MB
  __bf16* WhT_bf = (__bf16*)alloc((size_t)Bn * Hn * OUT * Nn * 2);  // 67 MB
  __bf16* e_bf   = (__bf16*)alloc((size_t)Bn * Hn * Nn * OUT * 2);  // 67 MB (reused as cat)
  __bf16* Am_bf  = (__bf16*)alloc((size_t)Bn * Hn * Nn * Nn * 2);   // 134 MB
  float*  hf_f   = (float*) alloc((size_t)Bn * Nn * OUT * 4);       // 33.5 MB
  float*  ha_f   = (float*) alloc((size_t)Bn * Nn * OUT * 4);       // 33.5 MB
  __bf16* fha    = (__bf16*)alloc((size_t)Bn * Nn * 2 * OUT * 2);   // 33.5 MB
  __bf16* cat_bf = e_bf;  // e dead after scores; hp writes concat layout here

  k_cvt  <<<4096, 256, 0, stream>>>(h,    h_bf,  (long long)Bn * Nn * IN);
  k_cvt  <<<1024, 256, 0, stream>>>(W,    W_bf,  (long long)Hn * OUT * IN);
  k_attnT<<<(Hn * OUT * OUT) / 256, 256, 0, stream>>>(attn, at_bf);
  k_cvt  <<<1024, 256, 0, stream>>>(A_w,  Aw_bf, (long long)OUT * Hn * OUT);
  k_cvt  <<<512,  256, 0, stream>>>(fc_w, fc_bf, (long long)OUT * IN);
  k_gcat <<<(OUT * 2 * OUT) / 256, 256, 0, stream>>>(g1, g2, gc_bf);

  // 1) Wh[b,h,n,o] = h @ W_h^T + bW_h      [M=1024, N=512, K=512] x64
  { GemmP p{};
    p.A = h_bf;  p.sAb = (long long)Nn * IN; p.sAh = 0; p.lda = IN;
    p.B = W_bf;  p.sBb = 0; p.sBh = (long long)OUT * IN; p.ldb = IN;
    p.Cb = Wh_bf; p.sCbB = (long long)Hn * Nn * OUT; p.sCbH = (long long)Nn * OUT; p.ldcb = OUT;
    p.bias = bW; p.sBiasH = OUT;
    p.M = Nn; p.Ncols = OUT; p.K = IN; p.H = Hn; p.epi = EPI_BIAS;
    gat_gemm<<<dim3(OUT / BN, Nn / BM, Bn * Hn), THREADS, 0, stream>>>(p); }

  // 1b) WhT[z][o][m] = Wh[z][m][o]   (for aggregation GEMM's B operand)
  k_transpose<<<dim3(OUT / 64, Nn / 64, Bn * Hn), 256, 0, stream>>>(Wh_bf, WhT_bf);

  // 2) e = Wh @ attn_h   (attn pre-transposed -> n-major)
  { GemmP p{};
    p.A = Wh_bf; p.sAb = (long long)Hn * Nn * OUT; p.sAh = (long long)Nn * OUT; p.lda = OUT;
    p.B = at_bf; p.sBb = 0; p.sBh = (long long)OUT * OUT; p.ldb = OUT;
    p.Cb = e_bf; p.sCbB = (long long)Hn * Nn * OUT; p.sCbH = (long long)Nn * OUT; p.ldcb = OUT;
    p.M = Nn; p.Ncols = OUT; p.K = OUT; p.H = Hn; p.epi = EPI_STORE;
    gat_gemm<<<dim3(OUT / BN, Nn / BM, Bn * Hn), THREADS, 0, stream>>>(p); }

  // 3) A_mat = tanh(adj ? e @ Wh^T : 0)   [M=1024, N=1024, K=512] x64
  { GemmP p{};
    p.A = e_bf;  p.sAb = (long long)Hn * Nn * OUT; p.sAh = (long long)Nn * OUT; p.lda = OUT;
    p.B = Wh_bf; p.sBb = (long long)Hn * Nn * OUT; p.sBh = (long long)Nn * OUT; p.ldb = OUT;
    p.Cb = Am_bf; p.sCbB = (long long)Hn * Nn * Nn; p.sCbH = (long long)Nn * Nn; p.ldcb = Nn;
    p.mask = adj; p.sMaskB = (long long)Nn * Nn; p.ldm = Nn;
    p.M = Nn; p.Ncols = Nn; p.K = OUT; p.H = Hn; p.epi = EPI_MASK_TANH;
    gat_gemm<<<dim3(Nn / BN, Nn / BM, Bn * Hn), THREADS, 0, stream>>>(p); }

  // 4) hp = relu(A_mat @ Wh) via WhT (n-major), written into concat [B,N,H*O]
  { GemmP p{};
    p.A = Am_bf;  p.sAb = (long long)Hn * Nn * Nn; p.sAh = (long long)Nn * Nn; p.lda = Nn;
    p.B = WhT_bf; p.sBb = (long long)Hn * OUT * Nn; p.sBh = (long long)OUT * Nn; p.ldb = Nn;
    p.Cb = cat_bf; p.sCbB = (long long)Nn * Hn * OUT; p.sCbH = OUT; p.ldcb = Hn * OUT;
    p.M = Nn; p.Ncols = OUT; p.K = Nn; p.H = Hn; p.epi = EPI_RELU;
    gat_gemm<<<dim3(OUT / BN, Nn / BM, Bn * Hn), THREADS, 0, stream>>>(p); }

  // 5) hf = h @ fc_w^T  (f32 + bf16-into-[hf|ha] buffer)
  { GemmP p{};
    p.A = h_bf;  p.sAb = (long long)Nn * IN; p.sAh = 0; p.lda = IN;
    p.B = fc_bf; p.sBb = 0; p.sBh = 0; p.ldb = IN;
    p.Cf = hf_f; p.sCfB = (long long)Nn * OUT; p.ldcf = OUT;
    p.Cb = fha;  p.sCbB = (long long)Nn * 2 * OUT; p.sCbH = 0; p.ldcb = 2 * OUT;
    p.M = Nn; p.Ncols = OUT; p.K = IN; p.H = 1; p.epi = EPI_F32_BF16;
    gat_gemm<<<dim3(OUT / BN, Nn / BM, Bn), THREADS, 0, stream>>>(p); }

  // 6) ha = cat @ A_w^T   [K = 2048]
  { GemmP p{};
    p.A = cat_bf; p.sAb = (long long)Nn * Hn * OUT; p.sAh = 0; p.lda = Hn * OUT;
    p.B = Aw_bf;  p.sBb = 0; p.sBh = 0; p.ldb = Hn * OUT;
    p.Cf = ha_f;  p.sCfB = (long long)Nn * OUT; p.ldcf = OUT;
    p.Cb = fha + OUT; p.sCbB = (long long)Nn * 2 * OUT; p.sCbH = 0; p.ldcb = 2 * OUT;
    p.M = Nn; p.Ncols = OUT; p.K = Hn * OUT; p.H = 1; p.epi = EPI_F32_BF16;
    gat_gemm<<<dim3(OUT / BN, Nn / BM, Bn), THREADS, 0, stream>>>(p); }

  // 7) coeff = sigmoid([hf|ha] @ [g1|g2]^T + gbias);  out = hf*c + ha*(1-c)
  { GemmP p{};
    p.A = fha;   p.sAb = (long long)Nn * 2 * OUT; p.sAh = 0; p.lda = 2 * OUT;
    p.B = gc_bf; p.sBb = 0; p.sBh = 0; p.ldb = 2 * OUT;
    p.hf = hf_f; p.ha = ha_f; p.outp = out; p.gbias = gb;
    p.M = Nn; p.Ncols = OUT; p.K = 2 * OUT; p.H = 1; p.epi = EPI_GATE;
    gat_gemm<<<dim3(OUT / BN, Nn / BM, Bn), THREADS, 0, stream>>>(p); }
}
